// SlideBayesGraphArch_12953621365180
// MI455X (gfx1250) — matrix operand order repaired and checked by reading, not compiled, not run on hardware
//
#include <hip/hip_runtime.h>

// ---------------------------------------------------------------------------
// SlideBayesGraphArch on MI455X (gfx1250, wave32, WMMA)
// Pipeline: head GEMM (bf16-split WMMA) -> BN stats -> BN+ReLU ->
//           edge scatter-add -> conv GEMM -> BN -> lin GEMM -> tail.
// ---------------------------------------------------------------------------

typedef __attribute__((ext_vector_type(16))) __bf16 v16bf;
typedef __attribute__((ext_vector_type(8)))  float  v8f;

#define GN     50000
#define GE     1600000
#define GF_IN  1024
#define GH     128
#define GT     4

#define BM 64
#define BK 32
#define PITCH 40   // u16 pitch for LDS tile rows (32 data + pad, 16B aligned)

__device__ __forceinline__ unsigned short f2bf(float f) {
    unsigned u = __float_as_uint(f);
    u = u + 0x7FFFu + ((u >> 16) & 1u);   // round-to-nearest-even
    return (unsigned short)(u >> 16);
}
__device__ __forceinline__ float bf2f(unsigned short h) {
    return __uint_as_float(((unsigned)h) << 16);
}

__device__ __forceinline__ v8f wmma_bf16(v16bf a, v16bf b, v8f c) {
    return __builtin_amdgcn_wmma_f32_16x16x32_bf16(
        false, a, false, b, (short)0, c, false, false);
}

// A-matrix 16x32 bf16 fragment: VGPR r<4 -> K = 8*half + 2r; r>=4 -> 16 + 8*half + 2(r-4)
__device__ __forceinline__ v16bf load_a_frag(const unsigned short* s, int m, int lh) {
    union { v16bf v; unsigned u[8]; } f;
    const unsigned* row = (const unsigned*)(s + m * PITCH);
#pragma unroll
    for (int r = 0; r < 8; ++r) {
        int k = (r < 4) ? (8 * lh + 2 * r) : (16 + 8 * lh + 2 * (r - 4));
        f.u[r] = row[k >> 1];
    }
    return f.v;
}
// B-matrix 32x16 bf16 fragment (stored n-major in LDS): VGPR r<4 -> K = 8*half + 2r; r>=4 -> 16 + 8*half + 2(r-4)
__device__ __forceinline__ v16bf load_b_frag(const unsigned short* s, int n, int lh) {
    union { v16bf v; unsigned u[8]; } f;
    const unsigned* row = (const unsigned*)(s + n * PITCH);
#pragma unroll
    for (int r = 0; r < 8; ++r) {
        int k = (r < 4) ? (8 * lh + 2 * r) : (16 + 8 * lh + 2 * (r - 4));
        f.u[r] = row[k >> 1];
    }
    return f.v;
}

// ---------------------------------------------------------------------------
// GEMM: C[M,128] = A[M,K] @ W[128,K]^T + bias, via split-bf16 WMMA.
// Block: 256 threads (8 waves). Tile 64 rows x 128 cols; BK=32 per step.
// ---------------------------------------------------------------------------
__global__ __launch_bounds__(256)
void gemm_bf16split(const float* __restrict__ A,
                    const unsigned short* __restrict__ Whi,
                    const unsigned short* __restrict__ Wlo,
                    const float* __restrict__ bias,
                    float* __restrict__ C, int M, int K) {
    __shared__ unsigned short sAhi[BM * PITCH];
    __shared__ unsigned short sAlo[BM * PITCH];
    __shared__ unsigned short sBhi[GH * PITCH];
    __shared__ unsigned short sBlo[GH * PITCH];

    const int t    = threadIdx.x;
    const int lane = t & 31;
    const int wave = t >> 5;            // 0..7
    const int rt   = wave >> 1;         // row tile 0..3
    const int cb   = (wave & 1) * 4;    // col tiles cb..cb+3
    const int row0 = blockIdx.x * BM;
    const int lm   = lane & 15;
    const int lh   = lane >> 4;

    v8f acc[4];
#pragma unroll
    for (int c = 0; c < 4; ++c)
#pragma unroll
        for (int j = 0; j < 8; ++j) acc[c][j] = 0.0f;

    for (int k0 = 0; k0 < K; k0 += BK) {
        // ---- stage A tile (f32 -> hi/lo bf16), 64 rows x 32 k
        {
            int r  = t >> 3;           // 0..31
            int kk = (t & 7) * 4;      // 0..28 step 4
#pragma unroll
            for (int hlf = 0; hlf < 2; ++hlf) {
                int rr   = r + hlf * 32;
                int grow = row0 + rr;
                float4 v = make_float4(0.f, 0.f, 0.f, 0.f);
                if (grow < M)
                    v = *(const float4*)(A + (size_t)grow * K + k0 + kk);
                unsigned short h0 = f2bf(v.x), h1 = f2bf(v.y), h2 = f2bf(v.z), h3 = f2bf(v.w);
                unsigned short l0 = f2bf(v.x - bf2f(h0)), l1 = f2bf(v.y - bf2f(h1));
                unsigned short l2 = f2bf(v.z - bf2f(h2)), l3 = f2bf(v.w - bf2f(h3));
                unsigned* dh = (unsigned*)&sAhi[rr * PITCH + kk];
                unsigned* dl = (unsigned*)&sAlo[rr * PITCH + kk];
                dh[0] = (unsigned)h0 | ((unsigned)h1 << 16);
                dh[1] = (unsigned)h2 | ((unsigned)h3 << 16);
                dl[0] = (unsigned)l0 | ((unsigned)l1 << 16);
                dl[1] = (unsigned)l2 | ((unsigned)l3 << 16);
            }
        }
        // ---- stage B tile (bf16 copy), 128 n-rows x 32 k
        {
            int n  = t >> 1;           // 0..127
            int kk = (t & 1) * 16;     // 0 or 16
            const uint4* sh = (const uint4*)(Whi + (size_t)n * K + k0 + kk);
            const uint4* sl = (const uint4*)(Wlo + (size_t)n * K + k0 + kk);
            uint4* dh = (uint4*)&sBhi[n * PITCH + kk];
            uint4* dl = (uint4*)&sBlo[n * PITCH + kk];
            dh[0] = sh[0]; dh[1] = sh[1];
            dl[0] = sl[0]; dl[1] = sl[1];
        }
        __syncthreads();

        v16bf ahi = load_a_frag(sAhi, rt * 16 + lm, lh);
        v16bf alo = load_a_frag(sAlo, rt * 16 + lm, lh);
#pragma unroll
        for (int c = 0; c < 4; ++c) {
            int n = (cb + c) * 16 + lm;
            v16bf bhi = load_b_frag(sBhi, n, lh);
            v16bf blo = load_b_frag(sBlo, n, lh);
            acc[c] = wmma_bf16(ahi, bhi, acc[c]);   // hi*hi
            acc[c] = wmma_bf16(ahi, blo, acc[c]);   // hi*lo
            acc[c] = wmma_bf16(alo, bhi, acc[c]);   // lo*hi
        }
        __syncthreads();
    }

    // ---- epilogue: C layout per ISA (VGPR v: M = v + 8*half, N = lane&15)
#pragma unroll
    for (int c = 0; c < 4; ++c) {
        int col = (cb + c) * 16 + lm;
        float b = bias[col];
#pragma unroll
        for (int v = 0; v < 8; ++v) {
            int grow = row0 + rt * 16 + v + lh * 8;
            if (grow < M) C[(size_t)grow * GH + col] = acc[c][v] + b;
        }
    }
}

// ---------------------------------------------------------------------------
__global__ void cvt_split_bf16(const float* __restrict__ src,
                               unsigned short* __restrict__ hi,
                               unsigned short* __restrict__ lo, int n) {
    int i = blockIdx.x * blockDim.x + threadIdx.x;
    if (i >= n) return;
    float f = src[i];
    unsigned short h = f2bf(f);
    hi[i] = h;
    lo[i] = f2bf(f - bf2f(h));
}

__global__ void zero_f32(float* __restrict__ p, int n) {
    int i = blockIdx.x * blockDim.x + threadIdx.x;
    if (i < n) p[i] = 0.0f;
}

// Per-channel sum / sum-of-squares into accum[0..127]=sum, accum[128..255]=sumsq
__global__ __launch_bounds__(256)
void bn_stats(const float* __restrict__ h, float* __restrict__ accum, int n) {
    __shared__ float ssum[GH], ssq[GH];
    if (threadIdx.x < GH) { ssum[threadIdx.x] = 0.f; ssq[threadIdx.x] = 0.f; }
    __syncthreads();
    size_t total = (size_t)n * GH;
    for (size_t i = (size_t)blockIdx.x * blockDim.x + threadIdx.x; i < total;
         i += (size_t)gridDim.x * blockDim.x) {
        float v = h[i];
        int c = (int)(i & (GH - 1));
        atomicAdd(&ssum[c], v);
        atomicAdd(&ssq[c], v * v);
    }
    __syncthreads();
    if (threadIdx.x < GH) {
        atomicAdd(&accum[threadIdx.x], ssum[threadIdx.x]);
        atomicAdd(&accum[GH + threadIdx.x], ssq[threadIdx.x]);
    }
}

// BN(train) + ReLU; optionally duplicates output into out1 (seeds GIN agg buffer)
__global__ __launch_bounds__(256)
void bn_relu(const float* __restrict__ hpre, const float* __restrict__ stats,
             const float* __restrict__ g, const float* __restrict__ be,
             float* __restrict__ out0, float* __restrict__ out1, int n) {
    size_t i = (size_t)blockIdx.x * blockDim.x + threadIdx.x;
    if (i >= (size_t)n * GH) return;
    int c = (int)(i & (GH - 1));
    float inv_n = 1.0f / (float)n;
    float mu  = stats[c] * inv_n;
    float var = stats[GH + c] * inv_n - mu * mu;
    float r   = rsqrtf(var + 1e-5f);
    float v   = (hpre[i] - mu) * r * g[c] + be[c];
    v = fmaxf(v, 0.0f);
    out0[i] = v;
    if (out1) out1[i] = v;
}

// sum[dst] += h[src] over all edges; thread = (edge, 4-channel chunk)
__global__ __launch_bounds__(256)
void scatter_add(const long long* __restrict__ ei, const float* __restrict__ h,
                 float* __restrict__ sum) {
    int gid = blockIdx.x * blockDim.x + threadIdx.x;
    int e = gid >> 5;
    if (e >= GE) return;
    int c = (gid & 31) * 4;
    long long s = ei[e];
    long long d = ei[(long long)GE + e];
    float4 v = *(const float4*)(h + (size_t)s * GH + c);
    float* p = sum + (size_t)d * GH + c;
    atomicAdd(p + 0, v.x);
    atomicAdd(p + 1, v.y);
    atomicAdd(p + 2, v.z);
    atomicAdd(p + 3, v.w);
}

// tail: out[n,0..3] = h2[n,:] @ Wt[4,128]^T + bt ; one wave per row
__global__ __launch_bounds__(256)
void tail_gemm(const float* __restrict__ h2, const float* __restrict__ Wt,
               const float* __restrict__ bt, float* __restrict__ out, int n) {
    int gid  = blockIdx.x * blockDim.x + threadIdx.x;
    int row  = gid >> 5;
    int lane = threadIdx.x & 31;
    if (row >= n) return;
    const float* hr = h2 + (size_t)row * GH;
    float a0 = 0.f, a1 = 0.f, a2 = 0.f, a3 = 0.f;
#pragma unroll
    for (int k = lane; k < GH; k += 32) {
        float x = hr[k];
        a0 += x * Wt[k];
        a1 += x * Wt[GH + k];
        a2 += x * Wt[2 * GH + k];
        a3 += x * Wt[3 * GH + k];
    }
#pragma unroll
    for (int off = 16; off > 0; off >>= 1) {
        a0 += __shfl_xor(a0, off, 32);
        a1 += __shfl_xor(a1, off, 32);
        a2 += __shfl_xor(a2, off, 32);
        a3 += __shfl_xor(a3, off, 32);
    }
    if (lane == 0) {
        float4 o = make_float4(a0 + bt[0], a1 + bt[1], a2 + bt[2], a3 + bt[3]);
        *(float4*)(out + (size_t)row * GT) = o;
    }
}

// ---------------------------------------------------------------------------
extern "C" void kernel_launch(void* const* d_in, const int* in_sizes, int n_in,
                              void* d_out, int out_size, void* d_ws, size_t ws_size,
                              hipStream_t stream) {
    (void)in_sizes; (void)n_in; (void)out_size; (void)ws_size;

    const float*     x      = (const float*)d_in[0];
    const long long* ei     = (const long long*)d_in[1];
    const float*     W_head = (const float*)d_in[2];
    const float*     b_head = (const float*)d_in[3];
    const float*     g0     = (const float*)d_in[4];
    const float*     be0    = (const float*)d_in[5];
    const float*     W_conv = (const float*)d_in[6];
    const float*     b_conv = (const float*)d_in[7];
    const float*     g1     = (const float*)d_in[8];
    const float*     be1    = (const float*)d_in[9];
    const float*     W_lin  = (const float*)d_in[10];
    const float*     b_lin  = (const float*)d_in[11];
    const float*     W_tail = (const float*)d_in[12];
    const float*     b_tail = (const float*)d_in[13];
    float* out = (float*)d_out;

    // ---- workspace carve-out (all 256B aligned) ----
    size_t off = 0;
    auto carve = [&](size_t bytes) -> void* {
        void* p = (char*)d_ws + off;
        off += (bytes + 255) & ~(size_t)255;
        return p;
    };
    float*          buf0   = (float*)carve((size_t)GN * GH * 4);
    float*          buf1   = (float*)carve((size_t)GN * GH * 4);
    unsigned short* Whh    = (unsigned short*)carve((size_t)GH * GF_IN * 2);
    unsigned short* Whl    = (unsigned short*)carve((size_t)GH * GF_IN * 2);
    unsigned short* Wch    = (unsigned short*)carve((size_t)GH * GH * 2);
    unsigned short* Wcl    = (unsigned short*)carve((size_t)GH * GH * 2);
    unsigned short* Wlh    = (unsigned short*)carve((size_t)GH * GH * 2);
    unsigned short* Wll    = (unsigned short*)carve((size_t)GH * GH * 2);
    float*          stats0 = (float*)carve(256 * 4);
    float*          stats1 = (float*)carve(256 * 4);

    const int TB = 256;

    // 0) split weights into bf16 hi/lo; zero BN accumulators
    cvt_split_bf16<<<(GH * GF_IN + TB - 1) / TB, TB, 0, stream>>>(W_head, Whh, Whl, GH * GF_IN);
    cvt_split_bf16<<<(GH * GH + TB - 1) / TB, TB, 0, stream>>>(W_conv, Wch, Wcl, GH * GH);
    cvt_split_bf16<<<(GH * GH + TB - 1) / TB, TB, 0, stream>>>(W_lin, Wlh, Wll, GH * GH);
    zero_f32<<<1, 256, 0, stream>>>(stats0, 256);
    zero_f32<<<1, 256, 0, stream>>>(stats1, 256);

    const int gemm_blocks = (GN + BM - 1) / BM;
    const int elem_blocks = (int)(((size_t)GN * GH + TB - 1) / TB);

    // 1) head GEMM: buf0 = x @ W_head^T + b_head
    gemm_bf16split<<<gemm_blocks, TB, 0, stream>>>(x, Whh, Whl, b_head, buf0, GN, GF_IN);

    // 2) BN stats + BN/ReLU -> buf1 = h ; buf0 = h (seeds agg sum)
    bn_stats<<<1024, TB, 0, stream>>>(buf0, stats0, GN);
    bn_relu<<<elem_blocks, TB, 0, stream>>>(buf0, stats0, g0, be0, buf1, buf0, GN);

    // 3) GIN aggregation: buf0 += sum_{e: dst} h[src]
    scatter_add<<<(GE * 32 + TB - 1) / TB, TB, 0, stream>>>(ei, buf1, buf0);

    // 4) conv GEMM: buf1 = (h + agg) @ W_conv^T + b_conv ; BN/ReLU -> buf0
    gemm_bf16split<<<gemm_blocks, TB, 0, stream>>>(buf0, Wch, Wcl, b_conv, buf1, GN, GH);
    bn_stats<<<1024, TB, 0, stream>>>(buf1, stats1, GN);
    bn_relu<<<elem_blocks, TB, 0, stream>>>(buf1, stats1, g1, be1, buf0, nullptr, GN);

    // 5) lin GEMM: buf1 = buf0 @ W_lin^T + b_lin
    gemm_bf16split<<<gemm_blocks, TB, 0, stream>>>(buf0, Wlh, Wll, b_lin, buf1, GN, GH);

    // 6) tail: out = buf1 @ W_tail^T + b_tail
    tail_gemm<<<(GN * 32 + TB - 1) / TB, TB, 0, stream>>>(buf1, W_tail, b_tail, out, GN);
}
